// ByteRingModel_7224134992270
// MI455X (gfx1250) — matrix operand order repaired and checked by reading, not compiled, not run on hardware
//
#include <hip/hip_runtime.h>
#include <hip/hip_bf16.h>
#include <stdint.h>

// Model dims (fixed by the reference)
#define PP   64    // ring positions
#define DD   32    // embedding dim
#define TT   512   // timesteps
#define BB   256   // batch
#define KIN  8     // input/output feature width

typedef __attribute__((ext_vector_type(16))) _Float16 v16h;
typedef __attribute__((ext_vector_type(8)))  float    v8f;

// ---------------------------------------------------------------------------
// Phase 1: sequential recurrence. One wave32 per batch; lane == embedding dim.
// Ring state + jump table + the whole x[b] sequence live in LDS.
// x[b] (16 KB) is staged with CDNA5 async global->LDS DMA (ASYNCcnt path).
// Emits s[b,t,:] (fp32) to workspace for the WMMA output projection.
// ---------------------------------------------------------------------------
__global__ __launch_bounds__(32) void ring_recurrence_kernel(
    const float* __restrict__ x,      // [B,T,8]
    const float* __restrict__ ptr0,   // [B]
    const float* __restrict__ Wi,     // [8,32]
    const float* __restrict__ bi,     // [32]
    const float* __restrict__ jump,   // [64]
    const float* __restrict__ Wg,     // [32]
    const float* __restrict__ bgp,    // [1]
    const float* __restrict__ csp,    // [1]
    float* __restrict__ S)            // [B*T, 32]
{
    __shared__ float ring[PP * DD];   // 8 KB
    __shared__ float xs[TT * KIN];    // 16 KB
    __shared__ float jmp[PP];         // 256 B

    const int lane = threadIdx.x;     // 0..31 == dim d
    const int b    = blockIdx.x;

    // ---- zero ring, stage jump table (intra-wave LDS is in-order) ----
    #pragma unroll
    for (int i = 0; i < PP * DD / 32; ++i) ring[i * 32 + lane] = 0.0f;
    jmp[lane]      = jump[lane];
    jmp[lane + 32] = jump[lane + 32];

    // ---- async DMA: 16 KB of x[b] -> LDS (32 lanes x 16 B x 32 iters) ----
    {
        uint32_t lds0 = (uint32_t)(uintptr_t)(&xs[0]);
        uint64_t g0   = (uint64_t)(uintptr_t)(x + (size_t)b * (TT * KIN));
        #pragma unroll 4
        for (int it = 0; it < 32; ++it) {
            uint32_t ldsa = lds0 + (uint32_t)((it * 32 + lane) * 16);
            uint64_t ga   = g0   + (uint64_t)((it * 32 + lane) * 16);
            asm volatile("global_load_async_to_lds_b128 %0, %1, off"
                         :: "v"(ldsa), "v"(ga) : "memory");
        }
        asm volatile("s_wait_asynccnt 0" ::: "memory");
        asm volatile("s_wait_dscnt 0x0" ::: "memory");
    }

    // ---- per-lane constants ----
    float wi[KIN];
    #pragma unroll
    for (int i = 0; i < KIN; ++i) wi[i] = Wi[i * DD + lane];
    const float biv = bi[lane];
    const float wg  = Wg[lane];
    const float bg  = bgp[0];
    const float cs  = 1.0f / (1.0f + expf(-csp[0]));   // sigmoid(context_strength)

    float ptr = ptr0[b];
    float h   = 0.0f;
    float* Srow = S + ((size_t)b * TT) * DD + lane;

    for (int t = 0; t < TT; ++t) {
        // input projection: inp[d] = bi[d] + sum_i x[t,i] * Wi[i,d]
        const float4* xv = (const float4*)(&xs[t * KIN]);
        float4 xa = xv[0], xb = xv[1];
        float inp = biv;
        inp = fmaf(xa.x, wi[0], inp); inp = fmaf(xa.y, wi[1], inp);
        inp = fmaf(xa.z, wi[2], inp); inp = fmaf(xa.w, wi[3], inp);
        inp = fmaf(xb.x, wi[4], inp); inp = fmaf(xb.y, wi[5], inp);
        inp = fmaf(xb.z, wi[6], inp); inp = fmaf(xb.w, wi[7], inp);

        // Gaussian attention window around the float pointer (wave-uniform)
        int base = (int)floorf(ptr);
        base = min(max(base, 0), PP - 1);
        float w[5]; int id[5];
        float esum = 0.0f;
        #pragma unroll
        for (int j = 0; j < 5; ++j) {
            int p = (base + j - 2 + PP) & (PP - 1);
            id[j] = p;
            float a = (float)p - ptr + 32.0f;              // (-32, 95]
            a = a - 64.0f * floorf(a * (1.0f / 64.0f));    // python remainder -> [0,64)
            float delta = a - 32.0f;
            float e = expf(-(delta * delta) * 0.125f);     // TEMP = 8
            w[j] = e; esum += e;
        }
        float inv = 1.0f / esum;
        float ctx = 0.0f;
        #pragma unroll
        for (int j = 0; j < 5; ++j) {
            w[j] *= inv;
            ctx = fmaf(w[j], ring[id[j] * DD + lane], ctx);
        }

        float s = tanhf(fmaf(cs, ctx, inp) + h);

        // weighted scatter-add write-back (5 distinct positions, one lane per elem)
        #pragma unroll
        for (int j = 0; j < 5; ++j) {
            int a = id[j] * DD + lane;
            ring[a] = fmaf(w[j], s, ring[a]);
        }

        // gate: logit = s . Wg + bg  (wave32 butterfly reduction)
        float g = s * wg;
        #pragma unroll
        for (int m = 16; m >= 1; m >>= 1) g += __shfl_xor(g, m, 32);
        float logit = g + bg;

        float jt   = jmp[base];
        float walk = ptr + 1.0f;
        if (walk >= 64.0f) walk -= 64.0f;
        ptr = (logit > 0.0f) ? jt : walk;                  // hard STE gate (fwd)
        h = s;

        Srow[(size_t)t * DD] = s;                          // coalesced 128B/wave
    }
}

// ---------------------------------------------------------------------------
// Phase 2: out = S[131072,32] @ Wo[32,8] + bo, via V_WMMA_F32_16X16X32_F16.
// One wave per 16-row tile; K=32 == full D in a single WMMA; N cols 8..15 = 0.
// Fragment layouts per CDNA5 ISA (16-bit A 16x32, B 32x16):
//   lanes 0-15  : K in {0..7, 16..23};  lanes 16-31 : K in {8..15, 24..31}
// ---------------------------------------------------------------------------
__global__ __launch_bounds__(256) void out_proj_wmma_kernel(
    const float* __restrict__ S,      // [R=131072, 32]
    const float* __restrict__ Wo,     // [32, 8]
    const float* __restrict__ bo,     // [8]
    float* __restrict__ out)          // [R, 8]
{
    const int lane = threadIdx.x & 31;
    const int wave = threadIdx.x >> 5;
    const int tile = blockIdx.x * 8 + wave;     // 1024 blocks * 8 waves = 8192 tiles
    const int half = lane >> 4;                 // which K-half set this lane holds
    const int n    = lane & 15;                 // N column / A row-in-tile
    const int row  = tile * 16 + n;             // A-matrix row (both lane halves)

    // ---- A fragment: S[row][k] for this lane's K set, fp32 -> fp16 ----
    const float4* Sr = (const float4*)(S + (size_t)row * DD);
    float4 f0 = Sr[2 * half + 0];   // k = 8*half   .. 8*half+3
    float4 f1 = Sr[2 * half + 1];   // k = 8*half+4 .. 8*half+7
    float4 f2 = Sr[2 * half + 4];   // k = 16+8*half   ..
    float4 f3 = Sr[2 * half + 5];   // k = 16+8*half+4 ..

    v16h A;
    A[0]=(_Float16)f0.x;  A[1]=(_Float16)f0.y;  A[2]=(_Float16)f0.z;  A[3]=(_Float16)f0.w;
    A[4]=(_Float16)f1.x;  A[5]=(_Float16)f1.y;  A[6]=(_Float16)f1.z;  A[7]=(_Float16)f1.w;
    A[8]=(_Float16)f2.x;  A[9]=(_Float16)f2.y;  A[10]=(_Float16)f2.z; A[11]=(_Float16)f2.w;
    A[12]=(_Float16)f3.x; A[13]=(_Float16)f3.y; A[14]=(_Float16)f3.z; A[15]=(_Float16)f3.w;

    // ---- B fragment: column n of Wo (zero-pad cols 8..15), K = 8*half+e (e<8), 16+8*half+(e-8) ----
    v16h Bf;
    #pragma unroll
    for (int e = 0; e < 16; ++e) {
        float v = (n < 8) ? Wo[(8 * half + e + (e >= 8 ? 8 : 0)) * 8 + n] : 0.0f;
        Bf[e] = (_Float16)v;
    }

    v8f acc = {};
    // (neg_a, A, neg_b, B, c_mod, C, reuse_a, reuse_b)
    acc = __builtin_amdgcn_wmma_f32_16x16x32_f16(
        false, A, false, Bf, (short)0, acc, false, false);

    // ---- D layout: VGPR r, lanes 0-15 -> (M=r, N=lane); 16-31 -> (M=r+8) ----
    if (n < 8) {
        const float bon = bo[n];
        float* orow = out + ((size_t)tile * 16) * KIN + n;
        #pragma unroll
        for (int r = 0; r < 8; ++r) {
            int M = r + 8 * half;
            orow[(size_t)M * KIN] = acc[r] + bon;
        }
    }
}

// ---------------------------------------------------------------------------
// Launch: setup_inputs order:
//  0:x 1:pointer_init 2:Wi 3:bi 4:Wo 5:bo 6:jump_dest 7:Wg 8:bg 9:context_strength
// ---------------------------------------------------------------------------
extern "C" void kernel_launch(void* const* d_in, const int* in_sizes, int n_in,
                              void* d_out, int out_size, void* d_ws, size_t ws_size,
                              hipStream_t stream) {
    const float* x   = (const float*)d_in[0];
    const float* p0  = (const float*)d_in[1];
    const float* Wi  = (const float*)d_in[2];
    const float* bi  = (const float*)d_in[3];
    const float* Wo  = (const float*)d_in[4];
    const float* bo  = (const float*)d_in[5];
    const float* jd  = (const float*)d_in[6];
    const float* Wg  = (const float*)d_in[7];
    const float* bg  = (const float*)d_in[8];
    const float* cs  = (const float*)d_in[9];
    float* out = (float*)d_out;
    float* S   = (float*)d_ws;                       // needs 16 MB: [B*T, 32] fp32

    ring_recurrence_kernel<<<BB, 32, 0, stream>>>(x, p0, Wi, bi, jd, Wg, bg, cs, S);
    out_proj_wmma_kernel<<<(BB * TT / 16) / 8, 256, 0, stream>>>(S, Wo, bo, out);
}
